// peepLSTM_42322607734853
// MI455X (gfx1250) — matrix operand order repaired and checked by reading, compiled, not run on hardware
//
#include <hip/hip_runtime.h>
#include <hip/hip_bf16.h>

// MI455X / gfx1250, wave32. Peephole-style LSTM:
//   - input-side gate projections collapse to 3-entry lookup tables (x in {0,1,2})
//   - recurrence: 3 GEMMs [16x256]@[256x256] per workgroup per step via
//     v_wmma_f32_16x16x32_bf16, cell state in registers, exchanged via LDS.
//   - 16 workgroups x 512 threads (16 waves), batch-split, no cross-WG sync.

typedef __attribute__((ext_vector_type(16))) __bf16 v16bf;
typedef __attribute__((ext_vector_type(8)))  float  v8f;

#define B_TOT  256
#define T_LEN  1024
#define D_DIM  128
#define H_DIM  256
#define C_DIM  10
#define ROWS   16          // batch rows per workgroup
#define NWAVE  16          // 512 threads

// d_ws layout: [0, 12KB) gate tables f32 [4 gates][3 vals][256]
//              [16KB, 16KB+384KB) packed bf16 recurrent weights
#define WS_W_OFF 16384

// ---------------------------------------------------------------------------
// Prep 1: gate lookup tables  tab[g][v][h] = b_g[h] + emb[v,:]·W_gx[:,h]
//         (sigmoid pre-applied for the candidate gate g==3)
// ---------------------------------------------------------------------------
__global__ void gate_tab_kernel(const float* __restrict__ emb,
                                const float* Wix, const float* bi,
                                const float* Wfx, const float* bfv,
                                const float* Wox, const float* bo,
                                const float* Wcx, const float* bc,
                                float* __restrict__ tab) {
    int g = blockIdx.x / 3, v = blockIdx.x % 3, h = threadIdx.x;
    const float* W; const float* b;
    switch (g) {
        case 0:  W = Wix; b = bi;  break;
        case 1:  W = Wfx; b = bfv; break;
        case 2:  W = Wox; b = bo;  break;
        default: W = Wcx; b = bc;  break;
    }
    float acc = b[h];
    for (int d = 0; d < D_DIM; ++d) acc += emb[v * D_DIM + d] * W[d * H_DIM + h];
    if (g == 3) acc = 1.f / (1.f + __expf(-acc));
    tab[(g * 3 + v) * H_DIM + h] = acc;
}

// ---------------------------------------------------------------------------
// Prep 2: repack f32 recurrent weights into bf16 WMMA B-tile layout.
// B tile (32x16 KxN, bf16): lane (n = lane&15) holds K = kchunk*32 +
// (lane<16 ? 0 : 16) + e  for e = 0..15 (contiguous k order, matching the A tiles).
// Tile address: pw + ((gate*16 + ntile)*8 + kchunk)*512 + lane*16
// ---------------------------------------------------------------------------
__global__ void pack_w_kernel(const float* __restrict__ Wih,
                              const float* __restrict__ Wfh,
                              const float* __restrict__ Woh,
                              __bf16* __restrict__ pw) {
    int blk = blockIdx.x;               // 3*16*8 = 384 blocks, 32 threads each
    int gate = blk >> 7, rem = blk & 127;
    int ntile = rem >> 3, kchunk = rem & 7;
    int lane = threadIdx.x;
    const float* W = (gate == 0) ? Wih : (gate == 1) ? Wfh : Woh;
    int n     = ntile * 16 + (lane & 15);
    int kbase = kchunk * 32 + ((lane < 16) ? 0 : 16);
    __bf16* dst = pw + (((gate * 16 + ntile) * 8 + kchunk) * 32 + lane) * 16;
#pragma unroll
    for (int e = 0; e < 16; ++e)
        dst[e] = (__bf16)W[(kbase + e) * H_DIM + n];
}

// ---------------------------------------------------------------------------
// Main recurrence kernel. One block = 16 batch rows, 16 waves.
// Wave w owns columns [16w,16w+16) of all 3 gate GEMMs; cell state is an
// f32 16x16 D-layout tile in registers. A-matrix (bf16 cell state) is
// rebuilt in LDS each step as A tiles in contiguous k order (k = kchunk*32 + 16*(lane/16) + e).
// ---------------------------------------------------------------------------
__global__ __launch_bounds__(512)
void lstm_main(const int* __restrict__ x, const float* __restrict__ tab_g,
               const __bf16* __restrict__ pw, const float* __restrict__ Wph,
               const float* __restrict__ bp, float* __restrict__ out) {
    __shared__ float tabs[4 * 3 * H_DIM];                    // 12 KB
    __shared__ __align__(32) __bf16 packedA[8][32][16];      // 8 KB (A tiles, kchunk-major)
    __shared__ int   x_sh[ROWS];
    __shared__ float h_sh[ROWS][H_DIM];                      // 16 KB
    __shared__ float p_sh[ROWS][C_DIM];

    const int tid  = threadIdx.x;
    const int lane = tid & 31;
    const int w    = tid >> 5;                  // wave id: owns ntile = w
    const int row0 = blockIdx.x * ROWS;

    for (int i = tid; i < 4 * 3 * H_DIM; i += 512) tabs[i] = tab_g[i];
    {   // c0 = 0
        __bf16* pa = &packedA[0][0][0];
        for (int i = tid; i < 8 * 32 * 16; i += 512) pa[i] = (__bf16)0.f;
    }

    const int hcol  = w * 16 + (lane & 15);     // my output column (D layout n)
    const int mbase = (lane < 16) ? 0 : 8;      // D layout: m = mbase + r

    // A-scatter target for my column, used as K index next step
    const int kidx   = hcol;
    const int kchS   = kidx >> 5;
    const int klocal = kidx & 31;
    const int hi8    = (klocal >> 4) & 1;
    const int kb     = klocal - (hi8 << 4);
    const int eS     = kb;

    v8f c_reg = {};
    v8f h_out = {};

    __syncthreads();

    for (int t = 0; t < T_LEN; ++t) {
        if (tid < ROWS) x_sh[tid] = x[(row0 + tid) * T_LEN + t];
        __syncthreads();   // B1: prev-step A writes + x_sh visible

        v8f ai = {}, af = {}, ao = {};
#pragma unroll
        for (int kk = 0; kk < 8; ++kk) {
            v16bf a  = *(const v16bf*)(&packedA[kk][lane][0]);
            v16bf b0 = *(const v16bf*)(pw + ((0 * 16 + w) * 8 + kk) * 512 + lane * 16);
            v16bf b1 = *(const v16bf*)(pw + ((1 * 16 + w) * 8 + kk) * 512 + lane * 16);
            v16bf b2 = *(const v16bf*)(pw + ((2 * 16 + w) * 8 + kk) * 512 + lane * 16);
            ai = __builtin_amdgcn_wmma_f32_16x16x32_bf16(false, a, false, b0, (short)0, ai, false, false);
            af = __builtin_amdgcn_wmma_f32_16x16x32_bf16(false, a, false, b1, (short)0, af, false, false);
            ao = __builtin_amdgcn_wmma_f32_16x16x32_bf16(false, a, false, b2, (short)0, ao, false, false);
        }

        float cn[8];
#pragma unroll
        for (int r = 0; r < 8; ++r) {
            int   m  = mbase + r;
            int   xv = x_sh[m];
            float gi = tabs[(0 * 3 + xv) * H_DIM + hcol];
            float gf = tabs[(1 * 3 + xv) * H_DIM + hcol];
            float go = tabs[(2 * 3 + xv) * H_DIM + hcol];
            float gc = tabs[(3 * 3 + xv) * H_DIM + hcol];   // sigmoid pre-applied
            float it = 1.f / (1.f + __expf(-(ai[r] + gi)));
            float ft = 1.f / (1.f + __expf(-(af[r] + gf)));
            float ot = 1.f / (1.f + __expf(-(ao[r] + go)));
            float cv = gc * it + c_reg[r] * ft;
            cn[r]    = cv;
            c_reg[r] = cv;
            if (t == T_LEN - 1) h_out[r] = tanhf(cv) * ot;
        }
        __syncthreads();   // B2: all reads of packedA for this step done

        // scatter bf16 cell state into next step's A layout
#pragma unroll
        for (int r = 0; r < 8; ++r)
            packedA[kchS][(mbase + r) + 16 * hi8][eS] = (__bf16)cn[r];
    }

    // final hidden state -> LDS
#pragma unroll
    for (int r = 0; r < 8; ++r) h_sh[mbase + r][hcol] = h_out[r];
    __syncthreads();

    // tiny projection [16x256]@[256x10] + bias
    if (tid < ROWS * C_DIM) {
        int m = tid / C_DIM, c = tid % C_DIM;
        float acc = bp[c];
        for (int k = 0; k < H_DIM; ++k) acc += h_sh[m][k] * Wph[k * C_DIM + c];
        p_sh[m][c] = acc;
    }
    __syncthreads();

    // log_softmax per row
    if (tid < ROWS) {
        int m = tid;
        float mx = p_sh[m][0];
        for (int c = 1; c < C_DIM; ++c) mx = fmaxf(mx, p_sh[m][c]);
        float s = 0.f;
        for (int c = 0; c < C_DIM; ++c) s += __expf(p_sh[m][c] - mx);
        float lse = mx + __logf(s);
        for (int c = 0; c < C_DIM; ++c)
            out[(row0 + m) * C_DIM + c] = p_sh[m][c] - lse;
    }
}

// ---------------------------------------------------------------------------
extern "C" void kernel_launch(void* const* d_in, const int* in_sizes, int n_in,
                              void* d_out, int out_size, void* d_ws, size_t ws_size,
                              hipStream_t stream) {
    const int*   x   = (const int*)d_in[0];
    const float* emb = (const float*)d_in[1];
    const float* Wix = (const float*)d_in[2];
    const float* Wih = (const float*)d_in[3];
    const float* bi  = (const float*)d_in[4];
    const float* Wfx = (const float*)d_in[5];
    const float* Wfh = (const float*)d_in[6];
    const float* bfv = (const float*)d_in[7];
    const float* Wox = (const float*)d_in[8];
    const float* Woh = (const float*)d_in[9];
    const float* bo  = (const float*)d_in[10];
    const float* Wcx = (const float*)d_in[11];
    const float* bc  = (const float*)d_in[12];
    const float* Wph = (const float*)d_in[13];
    const float* bp  = (const float*)d_in[14];
    float*  out = (float*)d_out;
    float*  tab = (float*)d_ws;
    __bf16* pw  = (__bf16*)((char*)d_ws + WS_W_OFF);

    gate_tab_kernel<<<12, 256, 0, stream>>>(emb, Wix, bi, Wfx, bfv, Wox, bo, Wcx, bc, tab);
    pack_w_kernel<<<384, 32, 0, stream>>>(Wih, Wfh, Woh, pw);
    lstm_main<<<B_TOT / ROWS, 512, 0, stream>>>(x, tab, pw, Wph, bp, out);
}
